// MoE_layer_flux_10892037063036
// MI455X (gfx1250) — matrix operand order repaired and checked by reading, not compiled, not run on hardware
//
#include <hip/hip_runtime.h>
#include <hip/hip_bf16.h>

// ---------------- problem constants ----------------
constexpr int T = 4096;
constexpr int H = 2048;
constexpr int F = 8192;
constexpr int E = 8;
constexpr int FC = 256;   // F-chunk width for the fused fc1->gelu->fc2 pipeline

// ---------------- WMMA vector types ----------------
typedef __attribute__((ext_vector_type(16))) __bf16 v16bf;
typedef __attribute__((ext_vector_type(8)))  __bf16 v8bf;
typedef __attribute__((ext_vector_type(8)))  float  v8f;

// ---------------- helpers ----------------
static __device__ __forceinline__ unsigned short f2bf_bits(float f) {
  unsigned int u = __builtin_bit_cast(unsigned int, f);
  u += 0x7FFFu + ((u >> 16) & 1u);              // round-to-nearest-even
  return (unsigned short)(u >> 16);
}
static __device__ __forceinline__ __bf16 f2bf(float f) {
  unsigned short s = f2bf_bits(f);
  return __builtin_bit_cast(__bf16, s);
}
// Branch-free tanh on the native v_exp_f32 path.
static __device__ __forceinline__ float tanh_fast(float u) {
  float a = __builtin_fabsf(u);
  float e = __expf(-2.0f * a);                  // v_exp_f32
  float t = __fdividef(1.0f - e, 1.0f + e);     // rcp-based fast divide
  return __builtin_copysignf(t, u);
}
static __device__ __forceinline__ float gelu_tanh(float x) {
  float u = 0.7978845608028654f * (x + 0.044715f * x * x * x);
  return 0.5f * x * (1.0f + tanh_fast(u));
}
// B-fragment: 16 consecutive bf16 (two b128 loads, zero VALU).
static __device__ __forceinline__ v16bf load_b_frag_bf16(const unsigned short* __restrict__ p) {
  v8bf lo = *reinterpret_cast<const v8bf*>(p);
  v8bf hi = *reinterpret_cast<const v8bf*>(p + 8);
  return __builtin_shufflevector(lo, hi, 0,1,2,3,4,5,6,7,8,9,10,11,12,13,14,15);
}
// B-fragment fallback: 16 consecutive fp32, convert in regs.
static __device__ __forceinline__ v16bf load_b_frag_f32(const float* __restrict__ p) {
  const float4* q = reinterpret_cast<const float4*>(p);
  float4 q0 = q[0], q1 = q[1], q2 = q[2], q3 = q[3];
  v16bf b;
  b[0]  = f2bf(q0.x); b[1]  = f2bf(q0.y); b[2]  = f2bf(q0.z); b[3]  = f2bf(q0.w);
  b[4]  = f2bf(q1.x); b[5]  = f2bf(q1.y); b[6]  = f2bf(q1.z); b[7]  = f2bf(q1.w);
  b[8]  = f2bf(q2.x); b[9]  = f2bf(q2.y); b[10] = f2bf(q2.z); b[11] = f2bf(q2.w);
  b[12] = f2bf(q3.x); b[13] = f2bf(q3.y); b[14] = f2bf(q3.z); b[15] = f2bf(q3.w);
  return b;
}
// A-fragment from LDS. A 16x32 in this code's k order (not the ISA table order): lane m = lane&15;
// elems 0..15 -> K = 16*half + j (contiguous, matches B).
static __device__ __forceinline__ v16bf load_a_frag_lds(const unsigned short* __restrict__ base) {
  v8bf alo = *reinterpret_cast<const v8bf*>(base);
  v8bf ahi = *reinterpret_cast<const v8bf*>(base + 8);
  return __builtin_shufflevector(alo, ahi, 0,1,2,3,4,5,6,7,8,9,10,11,12,13,14,15);
}

// ---------------- kernel 0: fp32 -> bf16 weight pre-conversion --------------
__global__ void conv_w_bf16(const float* __restrict__ w, unsigned short* __restrict__ o,
                            long long n4) {
  long long i = (long long)blockIdx.x * blockDim.x + threadIdx.x;
  long long stride = (long long)gridDim.x * blockDim.x;
  for (; i < n4; i += stride) {
    float4 v = reinterpret_cast<const float4*>(w)[i];
    ushort4 s;
    s.x = f2bf_bits(v.x); s.y = f2bf_bits(v.y); s.z = f2bf_bits(v.z); s.w = f2bf_bits(v.w);
    reinterpret_cast<ushort4*>(o)[i] = s;
  }
}

// ---------------- kernel 1: router (logits -> top2 -> softmax over k) --------
__global__ void moe_router(const float* __restrict__ x, const float* __restrict__ rw,
                           float* __restrict__ gates) {
  int t = blockIdx.x * blockDim.x + threadIdx.x;
  if (t >= T) return;
  float acc[E] = {};
  const float* xr = x + (size_t)t * H;
  for (int k = 0; k < H; ++k) {
    float xv = xr[k];
    const float* rwk = rw + (size_t)k * E;
#pragma unroll
    for (int e = 0; e < E; ++e) acc[e] += xv * rwk[e];
  }
  int i1 = 0; float v1 = acc[0];
#pragma unroll
  for (int e = 1; e < E; ++e) if (acc[e] > v1) { v1 = acc[e]; i1 = e; }
  int i2 = -1; float v2 = -__builtin_inff();
#pragma unroll
  for (int e = 0; e < E; ++e) if (e != i1 && acc[e] > v2) { v2 = acc[e]; i2 = e; }
  float p2  = __expf(v2 - v1);
  float inv = 1.0f / (1.0f + p2);
  float* g = gates + (size_t)t * E;
#pragma unroll
  for (int e = 0; e < E; ++e) g[e] = 0.0f;
  g[i1] = inv;
  g[i2] = p2 * inv;
}

// ---------------- kernel 2: per-expert ordered token lists -------------------
__global__ void moe_build_lists(const float* __restrict__ gates, int* __restrict__ counts,
                                int* __restrict__ toks) {
  int e = threadIdx.x;
  if (e >= E) return;
  int c = 0;
  for (int t = 0; t < T; ++t)
    if (gates[(size_t)t * E + e] > 0.0f) toks[(size_t)e * T + (c++)] = t;
  counts[e] = c;
}

// ---------------- kernel 3: fused grouped MLP (fc1 -> gelu -> fc2) -----------
// grid = (T/16, E); block = 256 threads = 8 waves.
// PRE = true: weights already bf16 in scratch; false: convert fp32 in regs.
template <bool PRE>
__global__ __launch_bounds__(256, 1)
void moe_expert_mlp(const float* __restrict__ x,
                    const void* __restrict__ w1p,
                    const void* __restrict__ w2p,
                    const float* __restrict__ gates,
                    const int* __restrict__ counts,
                    const int* __restrict__ toks,
                    float* __restrict__ out) {
  const int e  = blockIdx.y;
  const int m0 = blockIdx.x * 16;
  const int cnt = counts[e];
  if (m0 >= cnt) return;                       // block-uniform exit

  extern __shared__ unsigned char smem[];
  unsigned short* xs   = reinterpret_cast<unsigned short*>(smem);                 // [16][H]
  unsigned short* acts = reinterpret_cast<unsigned short*>(smem + 16 * H * 2);    // [16][FC]
  int*   row_tok  = reinterpret_cast<int*>(smem + 16 * H * 2 + 16 * FC * 2);      // [16]
  float* row_gate = reinterpret_cast<float*>(row_tok + 16);                       // [16]

  const int tid  = threadIdx.x;
  const int lane = tid & 31;
  const int wave = tid >> 5;
  const int half = lane >> 4;
  const int ln   = lane & 15;

  if (tid < 16) {
    int idx = m0 + tid;
    if (idx < cnt) {
      int tok = toks[(size_t)e * T + idx];
      row_tok[tid]  = tok;
      row_gate[tid] = gates[(size_t)tok * E + e];
    } else {
      row_tok[tid]  = toks[(size_t)e * T];     // valid address; gate 0 kills row
      row_gate[tid] = 0.0f;
    }
  }
  __syncthreads();

  // Stage x tile -> LDS bf16 (coalesced float4 reads, 8B LDS stores).
  for (int i = tid; i < (16 * H) / 4; i += 256) {
    int flat = i * 4;
    int r = flat >> 11;            // H == 2048
    int c = flat & (H - 1);
    float4 v = *reinterpret_cast<const float4*>(x + (size_t)row_tok[r] * H + c);
    ushort4 s;
    s.x = f2bf_bits(v.x); s.y = f2bf_bits(v.y); s.z = f2bf_bits(v.z); s.w = f2bf_bits(v.w);
    *reinterpret_cast<ushort4*>(xs + r * H + c) = s;
  }
  __syncthreads();

  v8f o[16] = {};                              // fc2 accumulators: 16 H-tiles / wave

  for (int fc0 = 0; fc0 < F; fc0 += FC) {
    // ---- fc1: this wave produces act columns [wave*32, wave*32+32) of chunk ----
    v8f c0 = {}, c1 = {};
    const int f_lo = fc0 + wave * 32;
    const size_t w1row0 = (size_t)e * F * H + (size_t)(f_lo + ln) * H;
    const size_t w1row1 = w1row0 + (size_t)16 * H;
    for (int kk = 0; kk < H; kk += 32) {
      v16bf a = load_a_frag_lds(xs + ln * H + kk + 16 * half);
      const size_t koff = (size_t)kk + 16 * half;
      v16bf b0, b1;
      if constexpr (PRE) {
        const unsigned short* wb = (const unsigned short*)w1p;
        __builtin_prefetch(wb + w1row0 + koff + 64, 0, 0);    // global_prefetch_b8
        b0 = load_b_frag_bf16(wb + w1row0 + koff);
        b1 = load_b_frag_bf16(wb + w1row1 + koff);
      } else {
        const float* wf = (const float*)w1p;
        b0 = load_b_frag_f32(wf + w1row0 + koff);
        b1 = load_b_frag_f32(wf + w1row1 + koff);
      }
      c0 = __builtin_amdgcn_wmma_f32_16x16x32_bf16(false, a, false, b0, (short)0, c0, false, false);
      c1 = __builtin_amdgcn_wmma_f32_16x16x32_bf16(false, a, false, b1, (short)0, c1, false, false);
    }
    __syncthreads();   // prior chunk's fc2 reads of acts are done
#pragma unroll
    for (int r = 0; r < 8; ++r) {
      int m = r + 8 * half;                                   // D layout: m = r + 8*half
      acts[m * FC + wave * 32 + ln]      = f2bf_bits(gelu_tanh(c0[r]));
      acts[m * FC + wave * 32 + 16 + ln] = f2bf_bits(gelu_tanh(c1[r]));
    }
    __syncthreads();   // acts chunk visible to all waves

    // ---- fc2: accumulate out tiles, K = this F-chunk ----
    const size_t w2row = (size_t)e * H * F + (size_t)(wave * 256 + ln) * F + fc0;
    for (int ks = 0; ks < FC; ks += 32) {
      v16bf a = load_a_frag_lds(acts + ln * FC + ks + 16 * half);
      const size_t boff = w2row + ks + 16 * half;
#pragma unroll
      for (int t = 0; t < 16; ++t) {
        v16bf b;
        if constexpr (PRE) {
          // tile offsets t*16*F are compile-time immediates (fit 24-bit ioffset)
          b = load_b_frag_bf16((const unsigned short*)w2p + boff + (size_t)(t * 16) * F);
        } else {
          b = load_b_frag_f32((const float*)w2p + boff + (size_t)(t * 16) * F);
        }
        o[t] = __builtin_amdgcn_wmma_f32_16x16x32_bf16(false, a, false, b, (short)0, o[t], false, false);
      }
    }
  }

  // ---- scale by gate and combine (<=2 experts per token) ----
#pragma unroll
  for (int t = 0; t < 16; ++t) {
    int h = wave * 256 + t * 16 + ln;
#pragma unroll
    for (int r = 0; r < 8; ++r) {
      int m = r + 8 * half;
      float v = o[t][r] * row_gate[m];
      atomicAdd(&out[(size_t)row_tok[m] * H + h], v);
    }
  }
}

// ---------------- launcher ----------------
extern "C" void kernel_launch(void* const* d_in, const int* in_sizes, int n_in,
                              void* d_out, int out_size, void* d_ws, size_t ws_size,
                              hipStream_t stream) {
  const float* x  = (const float*)d_in[0];   // [T,H]
  const float* rw = (const float*)d_in[1];   // [H,E]
  const float* w1 = (const float*)d_in[2];   // [E,F,H]
  const float* w2 = (const float*)d_in[3];   // [E,H,F]
  float* out = (float*)d_out;                // [T,H]

  char* ws = (char*)d_ws;
  float* gates  = (float*)ws;                                      // T*E floats
  int*   counts = (int*)(ws + (size_t)T * E * sizeof(float));      // E ints
  int*   toks   = (int*)(ws + (size_t)T * E * sizeof(float) + 32); // E*T ints

  const size_t wElems   = (size_t)E * F * H;                 // per weight tensor
  const size_t wOff     = 1ull << 20;                        // 1MB: router scratch
  const size_t needWs   = wOff + 2 * wElems * sizeof(unsigned short);
  unsigned short* w1b = (unsigned short*)(ws + wOff);
  unsigned short* w2b = (unsigned short*)(ws + wOff + wElems * sizeof(unsigned short));
  const bool pre = (ws_size >= needWs);

  hipMemsetAsync(d_out, 0, (size_t)T * H * sizeof(float), stream);

  if (pre) {  // one streaming pass: fp32 weights -> bf16 in scratch
    long long n4 = (long long)(wElems / 4);
    conv_w_bf16<<<dim3(8192), dim3(256), 0, stream>>>(w1, w1b, n4);
    conv_w_bf16<<<dim3(8192), dim3(256), 0, stream>>>(w2, w2b, n4);
  }

  moe_router<<<dim3((T + 255) / 256), dim3(256), 0, stream>>>(x, rw, gates);
  moe_build_lists<<<dim3(1), dim3(32), 0, stream>>>(gates, counts, toks);

  size_t smem = (size_t)16 * H * 2 + (size_t)16 * FC * 2 + 16 * 4 + 16 * 4; // 73856 B
  if (pre) {
    moe_expert_mlp<true><<<dim3(T / 16, E), dim3(256), smem, stream>>>(
        x, (const void*)w1b, (const void*)w2b, gates, counts, toks, out);
  } else {
    moe_expert_mlp<false><<<dim3(T / 16, E), dim3(256), smem, stream>>>(
        x, (const void*)w1, (const void*)w2, gates, counts, toks, out);
  }
}
